// CyclicalAttention_36670430773466
// MI455X (gfx1250) — matrix operand order repaired and checked by reading, not compiled, not run on hardware
//
#include <hip/hip_runtime.h>

typedef _Float16 half_t;
typedef __attribute__((ext_vector_type(16))) _Float16 v16h;
typedef __attribute__((ext_vector_type(8)))  _Float16 v8h;
typedef __attribute__((ext_vector_type(8)))  float    v8f;
typedef __attribute__((ext_vector_type(4))) unsigned int v4u;
typedef __attribute__((ext_vector_type(8)))  int v8i;
typedef __attribute__((ext_vector_type(4)))  int v4i;

#define D_MODEL  1024
#define N_HEADS  16
#define HEAD_DIM 64
#define BATCH    2
#define SEQ      2048
#define NTOK     (BATCH * SEQ)

#if defined(__has_builtin)
#  if __has_builtin(__builtin_amdgcn_tensor_load_to_lds)
#    define HAVE_TDM 1
#  endif
#  if __has_builtin(__builtin_amdgcn_s_wait_tensorcnt)
#    define HAVE_TCNT 1
#  endif
#endif

// ---------------------------------------------------------------------------
// WMMA helpers (CDNA5: D = A(16x32 f16) * B(32x16 f16) + C(16x16 f32))
// ---------------------------------------------------------------------------
__device__ __forceinline__ v8f wmma16(v16h a, v16h b, v8f c) {
  return __builtin_amdgcn_wmma_f32_16x16x32_f16(false, a, false, b, (short)0, c,
                                                false, false);
}

// A-matrix fragment (16 rows x 32 K), rows of `base` K-contiguous, stride ld.
// ISA: lanes 0-15 -> row l, K = {0..7, 16..23}; lanes 16-31 -> K = {8..15, 24..31}
__device__ __forceinline__ v16h load_a_frag(const half_t* __restrict__ base,
                                            long ld, int lane) {
  const int l  = lane & 15;
  const int hi = (lane >> 4) & 1;
  const half_t* p = base + (long)l * ld + hi * 8;
  v8h lo = *(const v8h*)p;
  v8h hh = *(const v8h*)(p + 16);
  v16h r;
#pragma unroll
  for (int i = 0; i < 8; ++i) { r[i] = lo[i]; r[i + 8] = hh[i]; }
  return r;
}

// B-matrix fragment (32 K x 16 N) where `base` holds column n K-contiguous.
// ISA: lanes 0-15 -> column l, K = {0..7, 16..23}; lanes 16-31 -> K = {8..15, 24..31}
__device__ __forceinline__ v16h load_b_frag(const half_t* __restrict__ base,
                                            long ld, int lane) {
  const int l  = lane & 15;
  const int hi = (lane >> 4) & 1;
  const half_t* p = base + (long)l * ld + hi * 8;
  v8h lo = *(const v8h*)p;
  v8h hh = *(const v8h*)(p + 16);
  v16h r;
#pragma unroll
  for (int i = 0; i < 8; ++i) { r[i] = lo[i]; r[i + 8] = hh[i]; }
  return r;
}

// ---------------------------------------------------------------------------
// Tensor Data Mover: load a [tile1 x tile0] f16 tile (tile1==0 -> 1D burst)
// from global into LDS.  D# built per ISA ch.8: group0 {count, lds_addr,
// global_addr, type=2}; group1 {data_size=2B, tensor dims, tile dims, stride}.
// Issue from ONE wave only (EXEC is ignored by TDM).
// ---------------------------------------------------------------------------
__device__ __forceinline__ void tdm_load_f16(half_t* lds_dst, const half_t* gsrc,
                                             unsigned tile0, unsigned tile1,
                                             unsigned dim0, unsigned dim1,
                                             unsigned long long stride0, int lane) {
#if defined(HAVE_TDM)
  const unsigned long long ga = (unsigned long long)(const void*)gsrc;
  const unsigned lds = (unsigned)(unsigned long long)(const void*)lds_dst;
  v4u g0;
  g0[0] = 1u;                                   // count=1, user descriptor
  g0[1] = lds;                                  // lds_addr (bytes)
  g0[2] = (unsigned)ga;                         // global_addr[31:0]
  g0[3] = (unsigned)(ga >> 32) | (2u << 30);    // global_addr[56:32] | type=2
  v8i g1;
  g1[0] = (int)(1u << 16);                      // wg_mask=0, data_size=1 (2B)
  g1[1] = (int)((dim0 & 0xFFFFu) << 16);                          // dim0[15:0]
  g1[2] = (int)(((dim0 >> 16) & 0xFFFFu) | ((dim1 & 0xFFFFu) << 16));
  g1[3] = (int)(((dim1 >> 16) & 0xFFFFu) | ((tile0 & 0xFFFFu) << 16));
  g1[4] = (int)(tile1 & 0xFFFFu);               // tile_dim1 | tile_dim2=0
  g1[5] = (int)(unsigned)stride0;               // tensor_dim0_stride[31:0]
  g1[6] = (int)((stride0 >> 32) & 0xFFFFu);     // stride0[47:32] | dim1_stride=0
  g1[7] = 0;
  v4i gz4 = {0, 0, 0, 0};
  v8i gz8 = {0, 0, 0, 0, 0, 0, 0, 0};
  // amdgpu-toolchain (clang-23) 6-arg form
  __builtin_amdgcn_tensor_load_to_lds(g0, g1, gz4, gz4, gz8, 0);
#else
  // fallback: lane-cooperative synchronous copy (called by one wave)
  const unsigned t1 = (tile1 == 0) ? 1u : tile1;
  for (unsigned r = 0; r < t1; ++r)
    for (unsigned c = lane; c < tile0; c += 32)
      lds_dst[(long)r * tile0 + c] = gsrc[(long)r * stride0 + c];
#endif
}

__device__ __forceinline__ void wait_tensor_le2() {
#if defined(HAVE_TDM)
#  if defined(HAVE_TCNT)
  __builtin_amdgcn_s_wait_tensorcnt(2);
#  else
  asm volatile("s_wait_tensorcnt 0x2" ::: "memory");
#  endif
#endif
}
__device__ __forceinline__ void wait_tensor_0() {
#if defined(HAVE_TDM)
#  if defined(HAVE_TCNT)
  __builtin_amdgcn_s_wait_tensorcnt(0);
#  else
  asm volatile("s_wait_tensorcnt 0x0" ::: "memory");
#  endif
#endif
}

// ---------------------------------------------------------------------------
// fp32 -> fp16 conversion (grid-stride)
// ---------------------------------------------------------------------------
__global__ void cvt_f16_kernel(const float* __restrict__ in,
                               half_t* __restrict__ out, int n) {
  int i = blockIdx.x * blockDim.x + threadIdx.x;
  int stride = gridDim.x * blockDim.x;
  for (; i < n; i += stride) out[i] = (half_t)in[i];
}

// ---------------------------------------------------------------------------
// cycle embedding + sequence L2-normalize: cn[b,h,s]
// ---------------------------------------------------------------------------
__global__ void cycle_norm_kernel(const float* __restrict__ tf,   // [B,S,12]
                                  const float* __restrict__ wc,   // [H,12]
                                  const float* __restrict__ bc,   // [H]
                                  float* __restrict__ cn) {       // [B*H, S]
  const int bh = blockIdx.x;
  const int b  = bh >> 4;
  const int h  = bh & 15;
  float wch[12];
#pragma unroll
  for (int d = 0; d < 12; ++d) wch[d] = wc[h * 12 + d];
  const float bch = bc[h];
  float* row = cn + (long)bh * SEQ;
  const float* tfb = tf + (long)b * SEQ * 12;

  float ss = 0.f;
  for (int s = threadIdx.x; s < SEQ; s += blockDim.x) {
    const float* t = tfb + s * 12;
    float e = bch;
#pragma unroll
    for (int d = 0; d < 12; ++d) e += t[d] * wch[d];
    row[s] = e;
    ss += e * e;
  }
  __shared__ float red[256];
  red[threadIdx.x] = ss;
  __syncthreads();
  for (int off = 128; off > 0; off >>= 1) {
    if (threadIdx.x < off) red[threadIdx.x] += red[threadIdx.x + off];
    __syncthreads();
  }
  const float inv = 1.f / fmaxf(sqrtf(red[0]), 1e-12f);
  for (int s = threadIdx.x; s < SEQ; s += blockDim.x) row[s] *= inv;
}

// ---------------------------------------------------------------------------
// Fused QKV projection: Q/K = [b,h,s,d], V stored transposed [b,h,d,s].
// grid (M/128, N/128, 3), block 256 (8 waves, wave tile 32x64)
// ---------------------------------------------------------------------------
__global__ void qkv_proj_kernel(const half_t* __restrict__ xh,   // [4096,1024]
                                const half_t* __restrict__ wqh,  // [1024,1024]
                                const half_t* __restrict__ wkh,
                                const half_t* __restrict__ wvh,
                                const float* __restrict__ bq,
                                const float* __restrict__ bk,
                                const float* __restrict__ bv,
                                half_t* __restrict__ Qh,
                                half_t* __restrict__ Kh,
                                half_t* __restrict__ Vt) {
  const int z = blockIdx.z;
  const half_t* w    = (z == 0) ? wqh : (z == 1) ? wkh : wvh;
  const float*  bias = (z == 0) ? bq  : (z == 1) ? bk  : bv;
  half_t*       out  = (z == 0) ? Qh  : (z == 1) ? Kh  : Vt;

  const int lane = threadIdx.x & 31;
  const int wave = threadIdx.x >> 5;
  const int wm = wave & 3, wn = wave >> 2;
  const int m0 = blockIdx.x * 128 + wm * 32;
  const int n0 = blockIdx.y * 128 + wn * 64;

  v8f acc[2][4] = {};
  for (int k0 = 0; k0 < D_MODEL; k0 += 32) {
    if (k0 + 32 < D_MODEL) {  // prefetch next k-tile (global_prefetch_b8)
      __builtin_prefetch(xh + (long)m0 * D_MODEL + k0 + 32, 0, 3);
      __builtin_prefetch(w + (long)n0 * D_MODEL + k0 + 32, 0, 3);
    }
    v16h a[2], bfr[4];
    a[0] = load_a_frag(xh + (long)m0 * D_MODEL + k0, D_MODEL, lane);
    a[1] = load_a_frag(xh + (long)(m0 + 16) * D_MODEL + k0, D_MODEL, lane);
#pragma unroll
    for (int j = 0; j < 4; ++j)
      bfr[j] = load_b_frag(w + (long)(n0 + 16 * j) * D_MODEL + k0, D_MODEL, lane);
#pragma unroll
    for (int i = 0; i < 2; ++i)
#pragma unroll
      for (int j = 0; j < 4; ++j) acc[i][j] = wmma16(a[i], bfr[j], acc[i][j]);
  }

  const int col = lane & 15, hi = lane >> 4;
#pragma unroll
  for (int i = 0; i < 2; ++i)
#pragma unroll
    for (int j = 0; j < 4; ++j)
#pragma unroll
      for (int r = 0; r < 8; ++r) {
        const int m = m0 + 16 * i + r + 8 * hi;   // token
        const int n = n0 + 16 * j + col;          // feature
        const float v = acc[i][j][r] + bias[n];
        const int b = m >> 11, s = m & (SEQ - 1);
        const int h = n >> 6,  d = n & (HEAD_DIM - 1);
        long idx;
        if (z == 2)  // V transposed: [b,h,d,s]
          idx = (((long)(b * N_HEADS + h)) * HEAD_DIM + d) * SEQ + s;
        else         // [b,h,s,d]
          idx = (((long)(b * N_HEADS + h)) * SEQ + s) * HEAD_DIM + d;
        out[idx] = (half_t)v;
      }
}

// ---------------------------------------------------------------------------
// Flash attention with rank-1 cyclical bias.
// grid (S/64, H, B), block 128 (4 waves); each wave owns 16 query rows.
// K/V tiles for the whole block staged in LDS by the Tensor Data Mover,
// double-buffered: DMA of tile i+1 overlaps WMMA compute on tile i.
// ---------------------------------------------------------------------------
__global__ void flash_attn_kernel(const half_t* __restrict__ Qh,  // [b,h,s,d]
                                  const half_t* __restrict__ Kh,  // [b,h,s,d]
                                  const half_t* __restrict__ Vt,  // [b,h,d,s]
                                  const float* __restrict__ cn,   // [b*h, s]
                                  const float* __restrict__ cscale, // [h]
                                  half_t* __restrict__ Oh) {      // [b,s,h,d]
  const int b = blockIdx.z, h = blockIdx.y;
  const int lane = threadIdx.x & 31;
  const int wave = threadIdx.x >> 5;
  const int q0 = blockIdx.x * 64 + wave * 16;
  const long bh = (long)b * N_HEADS + h;

  const half_t* Qp = Qh + (bh * SEQ + q0) * HEAD_DIM;
  const half_t* Kp = Kh + bh * SEQ * HEAD_DIM;
  const half_t* Vp = Vt + bh * HEAD_DIM * SEQ;
  const float*  cb = cn + bh * SEQ;
  const float scale_h = cscale[h];
  const float sm_scale = 0.125f;  // 1/sqrt(64)

  __shared__ half_t Kt[2][32][HEAD_DIM];   // 8 KB: 32 keys x 64 dims, dbl-buffered
  __shared__ half_t Vl[2][HEAD_DIM][32];   // 8 KB: 64 dims x 32 keys
  __shared__ half_t Pbuf[4][16][40];       // per-wave P transpose tile
  half_t (*P)[40] = Pbuf[wave];

  const int col = lane & 15, hi = lane >> 4;

  // Q fragments: 16 rows x 64 dims = 2 A-frags
  v16h aq[2];
  aq[0] = load_a_frag(Qp, HEAD_DIM, lane);
  aq[1] = load_a_frag(Qp + 32, HEAD_DIM, lane);

  float cq[8];
#pragma unroll
  for (int r = 0; r < 8; ++r) cq[r] = cb[q0 + r + 8 * hi];

  v8f o[4] = {};
  float Mrow[8], Lrow[8];
#pragma unroll
  for (int r = 0; r < 8; ++r) { Mrow[r] = -1e30f; Lrow[r] = 0.f; }

  const int NT = SEQ / 32;
  // prologue: stage tile 0 into buffer 0 (K burst is contiguous 32*64 f16)
  if (wave == 0) {
    tdm_load_f16(&Kt[0][0][0], Kp, 32 * HEAD_DIM, 0, 1u << 20, 1, 0, lane);
    tdm_load_f16(&Vl[0][0][0], Vp, 32, HEAD_DIM, SEQ, 1u << 20, SEQ, lane);
  }

  for (int it = 0; it < NT; ++it) {
    const int cur = it & 1;
    const int k0 = it * 32;
    __syncthreads();  // everyone done consuming buffer cur^1 (iter it-1)
    if (it + 1 < NT) {
      if (wave == 0) {
        const int k1 = k0 + 32;
        tdm_load_f16(&Kt[cur ^ 1][0][0], Kp + (long)k1 * HEAD_DIM,
                     32 * HEAD_DIM, 0, 1u << 20, 1, 0, lane);
        tdm_load_f16(&Vl[cur ^ 1][0][0], Vp + k1, 32, HEAD_DIM, SEQ, 1u << 20,
                     SEQ, lane);
      }
      wait_tensor_le2();  // tile `it` landed; tile `it+1` still in flight
    } else {
      wait_tensor_0();
    }
    __syncthreads();  // buffer cur is ready for all waves

    // ---- scores: 16 q x 32 k (K fragments from LDS) ----
    v8f sc[2] = {};
#pragma unroll
    for (int j = 0; j < 2; ++j) {
      v16h bk0 = load_b_frag(&Kt[cur][16 * j][0], HEAD_DIM, lane);
      v16h bk1 = load_b_frag(&Kt[cur][16 * j][32], HEAD_DIM, lane);
      sc[j] = wmma16(aq[0], bk0, sc[j]);
      sc[j] = wmma16(aq[1], bk1, sc[j]);
    }
    const float ck0 = cb[k0 + col];
    const float ck1 = cb[k0 + 16 + col];

    float s0[8], s1[8], mnew[8], fac[8];
#pragma unroll
    for (int r = 0; r < 8; ++r) {
      s0[r] = sc[0][r] * sm_scale + scale_h * cq[r] * ck0;
      s1[r] = sc[1][r] * sm_scale + scale_h * cq[r] * ck1;
    }
    // row max across the 16 lanes of this half (rows differ between halves)
#pragma unroll
    for (int r = 0; r < 8; ++r) {
      float m = fmaxf(s0[r], s1[r]);
#pragma unroll
      for (int off = 1; off < 16; off <<= 1) m = fmaxf(m, __shfl_xor(m, off, 32));
      mnew[r] = fmaxf(Mrow[r], m);
      fac[r] = __expf(Mrow[r] - mnew[r]);
    }
#pragma unroll
    for (int r = 0; r < 8; ++r) {
      const float p0 = __expf(s0[r] - mnew[r]);
      const float p1 = __expf(s1[r] - mnew[r]);
      float rs = p0 + p1;
#pragma unroll
      for (int off = 1; off < 16; off <<= 1) rs += __shfl_xor(rs, off, 32);
      Lrow[r] = Lrow[r] * fac[r] + rs;
      Mrow[r] = mnew[r];
      P[r + 8 * hi][col]      = (half_t)p0;
      P[r + 8 * hi][col + 16] = (half_t)p1;
    }
    // rescale accumulator
#pragma unroll
    for (int n = 0; n < 4; ++n)
#pragma unroll
      for (int r = 0; r < 8; ++r) o[n][r] *= fac[r];

    // re-read P as an A-fragment (transpose through LDS; same-wave order)
    v16h ap;
    {
      const half_t* pp = &P[lane & 15][hi * 8];
      v8h lo = *(const v8h*)pp;
      v8h hh = *(const v8h*)(pp + 16);
#pragma unroll
      for (int i = 0; i < 8; ++i) { ap[i] = lo[i]; ap[i + 8] = hh[i]; }
    }
    // ---- O += P * V (V fragments from LDS) ----
#pragma unroll
    for (int n = 0; n < 4; ++n) {
      v16h bvf = load_b_frag(&Vl[cur][n * 16][0], 32, lane);
      o[n] = wmma16(ap, bvf, o[n]);
    }
  }

  // epilogue: normalize and store to token-major [b, s, h, d]
#pragma unroll
  for (int n = 0; n < 4; ++n)
#pragma unroll
    for (int r = 0; r < 8; ++r) {
      const float v = o[n][r] / Lrow[r];
      const int srow = q0 + r + 8 * hi;
      const int d = n * 16 + col;
      Oh[(((long)b * SEQ + srow) * N_HEADS + h) * HEAD_DIM + d] = (half_t)v;
    }
}

// ---------------------------------------------------------------------------
// Output projection: out[m,n] = Oh[m,:] . wo[n,:] + bo[n]  (fp32 store)
// ---------------------------------------------------------------------------
__global__ void out_proj_kernel(const half_t* __restrict__ Oh,   // [4096,1024]
                                const half_t* __restrict__ woh,  // [1024,1024]
                                const float* __restrict__ bo,
                                float* __restrict__ out) {       // [4096,1024]
  const int lane = threadIdx.x & 31;
  const int wave = threadIdx.x >> 5;
  const int wm = wave & 3, wn = wave >> 2;
  const int m0 = blockIdx.x * 128 + wm * 32;
  const int n0 = blockIdx.y * 128 + wn * 64;

  v8f acc[2][4] = {};
  for (int k0 = 0; k0 < D_MODEL; k0 += 32) {
    if (k0 + 32 < D_MODEL) {
      __builtin_prefetch(Oh + (long)m0 * D_MODEL + k0 + 32, 0, 3);
      __builtin_prefetch(woh + (long)n0 * D_MODEL + k0 + 32, 0, 3);
    }
    v16h a[2], bfr[4];
    a[0] = load_a_frag(Oh + (long)m0 * D_MODEL + k0, D_MODEL, lane);
    a[1] = load_a_frag(Oh + (long)(m0 + 16) * D_MODEL + k0, D_MODEL, lane);
#pragma unroll
    for (int j = 0; j < 4; ++j)
      bfr[j] = load_b_frag(woh + (long)(n0 + 16 * j) * D_MODEL + k0, D_MODEL, lane);
#pragma unroll
    for (int i = 0; i < 2; ++i)
#pragma unroll
      for (int j = 0; j < 4; ++j) acc[i][j] = wmma16(a[i], bfr[j], acc[i][j]);
  }

  const int col = lane & 15, hi = lane >> 4;
#pragma unroll
  for (int i = 0; i < 2; ++i)
#pragma unroll
    for (int j = 0; j < 4; ++j)
#pragma unroll
      for (int r = 0; r < 8; ++r) {
        const int m = m0 + 16 * i + r + 8 * hi;
        const int n = n0 + 16 * j + col;
        out[(long)m * D_MODEL + n] = acc[i][j][r] + bo[n];
      }
}

// ---------------------------------------------------------------------------
// host launcher
// ---------------------------------------------------------------------------
extern "C" void kernel_launch(void* const* d_in, const int* in_sizes, int n_in,
                              void* d_out, int out_size, void* d_ws, size_t ws_size,
                              hipStream_t stream) {
  (void)in_sizes; (void)n_in; (void)out_size; (void)ws_size;

  const float* x  = (const float*)d_in[0];
  const float* tf = (const float*)d_in[1];
  const float* wq = (const float*)d_in[2];
  const float* bq = (const float*)d_in[3];
  const float* wk = (const float*)d_in[4];
  const float* bk = (const float*)d_in[5];
  const float* wv = (const float*)d_in[6];
  const float* bv = (const float*)d_in[7];
  const float* wo = (const float*)d_in[8];
  const float* bo = (const float*)d_in[9];
  const float* wc = (const float*)d_in[10];
  const float* bc = (const float*)d_in[11];
  const float* cs = (const float*)d_in[12];

  char* ws = (char*)d_ws;
  half_t* xh  = (half_t*)(ws);                          // 8 MB
  half_t* wqh = (half_t*)(ws + (8L  << 20));            // 2 MB each
  half_t* wkh = (half_t*)(ws + (10L << 20));
  half_t* wvh = (half_t*)(ws + (12L << 20));
  half_t* woh = (half_t*)(ws + (14L << 20));
  half_t* Qh  = (half_t*)(ws + (16L << 20));            // 8 MB
  half_t* Kh  = (half_t*)(ws + (24L << 20));            // 8 MB
  half_t* Vt  = (half_t*)(ws + (32L << 20));            // 8 MB
  half_t* Oh  = (half_t*)(ws + (40L << 20));            // 8 MB
  float*  cn  = (float*) (ws + (48L << 20));            // 256 KB

  cvt_f16_kernel<<<512, 256, 0, stream>>>(x,  xh,  NTOK * D_MODEL);
  cvt_f16_kernel<<<256, 256, 0, stream>>>(wq, wqh, D_MODEL * D_MODEL);
  cvt_f16_kernel<<<256, 256, 0, stream>>>(wk, wkh, D_MODEL * D_MODEL);
  cvt_f16_kernel<<<256, 256, 0, stream>>>(wv, wvh, D_MODEL * D_MODEL);
  cvt_f16_kernel<<<256, 256, 0, stream>>>(wo, woh, D_MODEL * D_MODEL);

  cycle_norm_kernel<<<BATCH * N_HEADS, 256, 0, stream>>>(tf, wc, bc, cn);

  qkv_proj_kernel<<<dim3(NTOK / 128, D_MODEL / 128, 3), 256, 0, stream>>>(
      xh, wqh, wkh, wvh, bq, bk, bv, Qh, Kh, Vt);

  flash_attn_kernel<<<dim3(SEQ / 64, N_HEADS, BATCH), 128, 0, stream>>>(
      Qh, Kh, Vt, cn, cs, Oh);

  out_proj_kernel<<<dim3(NTOK / 128, D_MODEL / 128), 256, 0, stream>>>(
      Oh, woh, bo, (float*)d_out);
}
